// Model_74741020885362
// MI455X (gfx1250) — compile-verified, matrix operand order checked by reading, not run on hardware
//
#include <hip/hip_runtime.h>
#include <hip/hip_bf16.h>
#include <math.h>

// ---------------- problem constants (from reference) ----------------
#define FEATD   300
#define HIDD    512
#define CDIM    20
#define NLAYERS 2
#define NHEADS  4
#define KHOPS   3
#define ALPHAD  0.1f
#define NNODES  76800
#define NEDGES  1250000
#define NGRAPH  512
#define SLOPE   0.2f

// ---------------- WMMA tiling ----------------
#define KPAD 320              // FEAT padded to a multiple of 32
#define KT   (KPAD / 32)      // 10 k-tiles
#define MT   (NNODES / 16)    // 4800 m-tiles
#define NT   (HIDD / 16)      // 32 n-tiles
#define FRAG 512              // halves per 16x32 fragment (32 lanes * 16 halves)
#define MB   2                // m-tiles per wave (register blocking)
#define NB   4                // n-tiles per wave
#define NTB  (NT / NB)        // 8 n-blocks
#define MTB  (MT / MB)        // 2400 m-blocks

typedef __attribute__((ext_vector_type(16))) _Float16 v16h;
typedef __attribute__((ext_vector_type(8)))  float    v8f;

// K index inside a 16x32 f16 fragment for (lane, element i of v16h),
// per CDNA5 ISA 7.12.2 "16-bit A-Matrix 16x32":
//   lanes 0-15: VGPR0..3 -> K=0..7, VGPR4..7 -> K=16..23
//   lanes16-31: VGPR0..3 -> K=8..15, VGPR4..7 -> K=24..31
__device__ __forceinline__ int frag_k(int lane, int i) {
    int kbase = (lane < 16) ? 0 : 8;
    return ((i >> 3) << 4) + kbase + (((i >> 1) & 3) << 1) + (i & 1);
}

// ---------------- small utilities ----------------
__global__ void zero_f32(float* __restrict__ p, int n) {
    int i = blockIdx.x * blockDim.x + threadIdx.x;
    if (i < n) p[i] = 0.f;
}
__global__ void copy_f32(float* __restrict__ dst, const float* __restrict__ src, int n) {
    int i = blockIdx.x * blockDim.x + threadIdx.x;
    if (i < n) dst[i] = src[i];
}
__global__ void init_seg(float* __restrict__ maxb, float* __restrict__ denom, int n) {
    int i = blockIdx.x * blockDim.x + threadIdx.x;
    if (i < n) { maxb[i] = __int_as_float(0xFF800000); denom[i] = 0.f; }
}

// ordered float atomic-max: int max for >=0, uint min for <0 (monotone encodings)
__device__ __forceinline__ void atomicMaxF(float* addr, float val) {
    if (val >= 0.f) atomicMax((int*)addr, __float_as_int(val));
    else            atomicMin((unsigned int*)addr, __float_as_uint(val));
}

// ---------------- K0a: gather emb[node_ids] -> f16 A fragments ----------------
// Afrag layout: [mtile][kt][lane][16] halves, contiguous per lane.
__global__ void pack_A(const int* __restrict__ node_ids, const float* __restrict__ emb,
                       _Float16* __restrict__ Af) {
    int tid = blockIdx.x * blockDim.x + threadIdx.x;
    if (tid >= MT * KT * FRAG) return;
    int within = tid & (FRAG - 1);
    int fragid = tid >> 9;            // / FRAG
    int lane = within >> 4, i = within & 15;
    int kt = fragid % KT, mtile = fragid / KT;
    int m = mtile * 16 + (lane & 15);
    int kglob = kt * 32 + frag_k(lane, i);
    float v = 0.f;
    if (kglob < FEATD) v = emb[(size_t)node_ids[m] * FEATD + kglob];
    Af[tid] = (_Float16)v;
}

// ---------------- K0b: W1 -> f16 B fragments ----------------
// Bfrag layout: [kt][ntile][lane][16]; B column n = lane&15, K pattern symmetric to A.
__global__ void pack_B(const float* __restrict__ W1, _Float16* __restrict__ Bf) {
    int tid = blockIdx.x * blockDim.x + threadIdx.x;
    if (tid >= KT * NT * FRAG) return;
    int within = tid & (FRAG - 1);
    int fragid = tid >> 9;
    int lane = within >> 4, i = within & 15;
    int ntile = fragid % NT, kt = fragid / NT;
    int n = ntile * 16 + (lane & 15);
    int kglob = kt * 32 + frag_k(lane, i);
    float v = 0.f;
    if (kglob < FEATD) v = W1[(size_t)kglob * HIDD + n];
    Bf[tid] = (_Float16)v;
}

// ---------------- K1: x1 = relu(A @ W1 + b1), register-blocked WMMA ----------------
// Each wave computes a (MB*16) x (NB*16) output block: 8 accumulators, 8 WMMAs
// per k-step off 2 A-frag + 4 B-frag loads.
__global__ void gemm1_wmma(const _Float16* __restrict__ Af, const _Float16* __restrict__ Bf,
                           const float* __restrict__ b1, float* __restrict__ x1) {
    int wave = blockIdx.x * (blockDim.x >> 5) + (threadIdx.x >> 5);
    int lane = threadIdx.x & 31;
    int mblk = wave / NTB;
    int nblk = wave - mblk * NTB;
    if (mblk >= MTB) return;

    v8f c[MB][NB];
#pragma unroll
    for (int m = 0; m < MB; ++m)
#pragma unroll
        for (int n = 0; n < NB; ++n) c[m][n] = (v8f){};

    const _Float16* ap[MB];
    const _Float16* bp[NB];
#pragma unroll
    for (int m = 0; m < MB; ++m)
        ap[m] = Af + (size_t)((mblk * MB + m) * KT) * FRAG + lane * 16;
#pragma unroll
    for (int n = 0; n < NB; ++n)
        bp[n] = Bf + (size_t)(nblk * NB + n) * FRAG + lane * 16;

#pragma unroll
    for (int kt = 0; kt < KT; ++kt) {
        v16h a[MB], b[NB];
#pragma unroll
        for (int m = 0; m < MB; ++m) {
            if (kt + 1 < KT) __builtin_prefetch(ap[m] + FRAG, 0, 3);  // global_prefetch
            a[m] = *(const v16h*)ap[m];
            ap[m] += FRAG;
        }
#pragma unroll
        for (int n = 0; n < NB; ++n) {
            b[n] = *(const v16h*)bp[n];
            bp[n] += (size_t)NT * FRAG;
        }
#pragma unroll
        for (int m = 0; m < MB; ++m)
#pragma unroll
            for (int n = 0; n < NB; ++n)
                c[m][n] = __builtin_amdgcn_wmma_f32_16x16x32_f16(
                    false, a[m], false, b[n], (short)0, c[m][n], false, false);
    }

    // C/D layout: VGPR v: lanes 0-15 -> M=v, lanes 16-31 -> M=8+v; N = lane&15
#pragma unroll
    for (int m = 0; m < MB; ++m) {
        int mrow0 = (mblk * MB + m) * 16 + ((lane >> 4) << 3);
#pragma unroll
        for (int n = 0; n < NB; ++n) {
            int ncol = (nblk * NB + n) * 16 + (lane & 15);
            float bias = b1[ncol];
#pragma unroll
            for (int v = 0; v < 8; ++v) {
                float r = c[m][n][v] + bias;
                x1[(size_t)(mrow0 + v) * HIDD + ncol] = r > 0.f ? r : 0.f;
            }
        }
    }
}

// ---------------- K2: h = x1 @ W2 + b2 (VALU, W2 broadcast from LDS) ----------------
__global__ void gemm2(const float* __restrict__ x1, const float* __restrict__ W2,
                      const float* __restrict__ b2, float* __restrict__ h) {
    __shared__ float w[HIDD * CDIM];
    for (int i = threadIdx.x; i < HIDD * CDIM; i += blockDim.x) w[i] = W2[i];
    __syncthreads();
    int n = blockIdx.x * blockDim.x + threadIdx.x;
    if (n >= NNODES) return;
    float acc[CDIM];
#pragma unroll
    for (int c = 0; c < CDIM; ++c) acc[c] = b2[c];
    const float* xp = x1 + (size_t)n * HIDD;
    for (int j = 0; j < HIDD; j += 4) {
        float4 xv = *(const float4*)(xp + j);
#pragma unroll
        for (int c = 0; c < CDIM; ++c) {
            acc[c] += xv.x * w[(j + 0) * CDIM + c] + xv.y * w[(j + 1) * CDIM + c]
                    + xv.z * w[(j + 2) * CDIM + c] + xv.w * w[(j + 3) * CDIM + c];
        }
    }
#pragma unroll
    for (int c = 0; c < CDIM; ++c) h[(size_t)n * CDIM + c] = acc[c];
}

// ---------------- K3: per-head projection z + attention logits es/ed ----------------
__global__ void gat_prep(const float* __restrict__ h, const float* __restrict__ gatW,
                         const float* __restrict__ asrc, const float* __restrict__ adst,
                         float* __restrict__ z, float* __restrict__ es, float* __restrict__ ed) {
    __shared__ float w[NHEADS * CDIM * CDIM];
    __shared__ float sa[NHEADS * CDIM];
    __shared__ float sd[NHEADS * CDIM];
    for (int i = threadIdx.x; i < NHEADS * CDIM * CDIM; i += blockDim.x) w[i] = gatW[i];
    for (int i = threadIdx.x; i < NHEADS * CDIM; i += blockDim.x) { sa[i] = asrc[i]; sd[i] = adst[i]; }
    __syncthreads();
    int n = blockIdx.x * blockDim.x + threadIdx.x;
    if (n >= NNODES) return;
    float hr[CDIM];
#pragma unroll
    for (int c = 0; c < CDIM; ++c) hr[c] = h[(size_t)n * CDIM + c];
    for (int hd = 0; hd < NHEADS; ++hd) {
        float se = 0.f, de = 0.f;
#pragma unroll
        for (int d = 0; d < CDIM; ++d) {
            float acc = 0.f;
#pragma unroll
            for (int c = 0; c < CDIM; ++c) acc += hr[c] * w[(hd * CDIM + c) * CDIM + d];
            z[((size_t)hd * NNODES + n) * CDIM + d] = acc;
            se += acc * sa[hd * CDIM + d];
            de += acc * sd[hd * CDIM + d];
        }
        es[(size_t)hd * NNODES + n] = se;
        ed[(size_t)hd * NNODES + n] = de;
    }
}

// ---------------- K4: segment-max of leaky_relu edge scores (L2 atomics) ----------------
__global__ void edge_max(const int* __restrict__ src, const int* __restrict__ dst,
                         const float* __restrict__ es, const float* __restrict__ ed,
                         float* __restrict__ maxb) {
    int e = blockIdx.x * blockDim.x + threadIdx.x;
    if (e >= NEDGES) return;
    int s = src[e], d = dst[e];
#pragma unroll
    for (int hd = 0; hd < NHEADS; ++hd) {
        float v = es[(size_t)hd * NNODES + s] + ed[(size_t)hd * NNODES + d];
        v = v > 0.f ? v : SLOPE * v;
        atomicMaxF(&maxb[(size_t)hd * NNODES + d], v);
    }
}

__global__ void max_fix(float* __restrict__ maxb, int n) {
    int i = blockIdx.x * blockDim.x + threadIdx.x;
    if (i < n) { float m = maxb[i]; maxb[i] = (m > -3.0e38f) ? m : 0.f; }
}

// ---------------- K5: ex = exp(e - max[dst]); denom[dst] += ex ----------------
__global__ void edge_exp(const int* __restrict__ src, const int* __restrict__ dst,
                         const float* __restrict__ es, const float* __restrict__ ed,
                         const float* __restrict__ maxb, float* __restrict__ denom,
                         float* __restrict__ att) {
    int e = blockIdx.x * blockDim.x + threadIdx.x;
    if (e >= NEDGES) return;
    int s = src[e], d = dst[e];
#pragma unroll
    for (int hd = 0; hd < NHEADS; ++hd) {
        float v = es[(size_t)hd * NNODES + s] + ed[(size_t)hd * NNODES + d];
        v = v > 0.f ? v : SLOPE * v;
        float ex = __expf(v - maxb[(size_t)hd * NNODES + d]);
        att[(size_t)hd * NEDGES + e] = ex;
        atomicAdd(&denom[(size_t)hd * NNODES + d], ex);
    }
}

// ---------------- K6: att /= (denom[dst] + 1e-9) ----------------
__global__ void att_norm(const int* __restrict__ dst, const float* __restrict__ denom,
                         float* __restrict__ att) {
    int tid = blockIdx.x * blockDim.x + threadIdx.x;
    if (tid >= NHEADS * NEDGES) return;
    int hd = tid / NEDGES, e = tid - hd * NEDGES;
    att[tid] = att[tid] / (denom[(size_t)hd * NNODES + dst[e]] + 1e-9f);
}

// ---------------- K7b: agg[dst] += att * Hc[src]  (L2-resident scatter) ----------------
__global__ void diffuse(const int* __restrict__ src, const int* __restrict__ dst,
                        const float* __restrict__ att, const float* __restrict__ Hc,
                        float* __restrict__ agg) {
    int tid = blockIdx.x * blockDim.x + threadIdx.x;
    if (tid >= NHEADS * NEDGES) return;
    int hd = tid / NEDGES, e = tid - hd * NEDGES;
    int s = src[e], d = dst[e];
    float a = att[tid];
    const float* hp = Hc + ((size_t)hd * NNODES + s) * CDIM;
    float* op = agg + ((size_t)hd * NNODES + d) * CDIM;
#pragma unroll
    for (int c = 0; c < CDIM; ++c) atomicAdd(op + c, a * hp[c]);
}

// ---------------- K7c: Hc = (1-alpha)*agg + alpha*H0 ----------------
__global__ void combine(float* __restrict__ Hc, const float* __restrict__ agg,
                        const float* __restrict__ H0, int n) {
    int i = blockIdx.x * blockDim.x + threadIdx.x;
    if (i < n) Hc[i] = (1.f - ALPHAD) * agg[i] + ALPHAD * H0[i];
}

// ---------------- K8: mean over heads (+ optional ELU) ----------------
__global__ void head_mean(const float* __restrict__ Hc, float* __restrict__ h, int apply_elu) {
    int tid = blockIdx.x * blockDim.x + threadIdx.x;
    if (tid >= NNODES * CDIM) return;
    float s = 0.f;
#pragma unroll
    for (int hd = 0; hd < NHEADS; ++hd) s += Hc[(size_t)hd * NNODES * CDIM + tid];
    s *= (1.f / NHEADS);
    if (apply_elu) s = s > 0.f ? s : expm1f(s);
    h[tid] = s;
}

// ---------------- K9: gated readout into per-graph sums ----------------
__global__ void readout(const float* __restrict__ h, const int* __restrict__ graph_ids,
                        const float* __restrict__ wsW, const float* __restrict__ wsB,
                        float* __restrict__ out) {
    int n = blockIdx.x * blockDim.x + threadIdx.x;
    if (n >= NNODES) return;
    float hr[CDIM];
    float g = wsB[0];
#pragma unroll
    for (int c = 0; c < CDIM; ++c) { hr[c] = h[(size_t)n * CDIM + c]; g += hr[c] * wsW[c]; }
    g = 1.f / (1.f + __expf(-g));
    float* op = out + (size_t)graph_ids[n] * CDIM;
#pragma unroll
    for (int c = 0; c < CDIM; ++c) atomicAdd(op + c, hr[c] * g);
}

// ---------------- orchestration ----------------
extern "C" void kernel_launch(void* const* d_in, const int* in_sizes, int n_in,
                              void* d_out, int out_size, void* d_ws, size_t ws_size,
                              hipStream_t stream) {
    (void)in_sizes; (void)n_in; (void)out_size; (void)ws_size;
    const int*   node_ids  = (const int*)d_in[0];
    const int*   edge_src  = (const int*)d_in[1];
    const int*   edge_dst  = (const int*)d_in[2];
    const int*   graph_ids = (const int*)d_in[3];
    const float* emb       = (const float*)d_in[4];
    const float* W1        = (const float*)d_in[5];
    const float* b1        = (const float*)d_in[6];
    const float* W2        = (const float*)d_in[7];
    const float* b2        = (const float*)d_in[8];
    const float* gat_W     = (const float*)d_in[9];
    const float* gat_asrc  = (const float*)d_in[10];
    const float* gat_adst  = (const float*)d_in[11];
    const float* ws_W      = (const float*)d_in[12];
    const float* ws_b      = (const float*)d_in[13];
    float* out = (float*)d_out;

    char* base = (char*)d_ws;
    size_t off = 0;
    auto carve = [&](size_t bytes) -> void* {
        void* p = base + off;
        off = (off + bytes + 255) & ~(size_t)255;
        return p;
    };
    _Float16* Af   = (_Float16*)carve((size_t)MT * KT * FRAG * sizeof(_Float16));
    _Float16* Bf   = (_Float16*)carve((size_t)KT * NT * FRAG * sizeof(_Float16));
    float* x1      = (float*)carve((size_t)NNODES * HIDD * sizeof(float));
    float* hbuf    = (float*)carve((size_t)NNODES * CDIM * sizeof(float));
    float* z       = (float*)carve((size_t)NHEADS * NNODES * CDIM * sizeof(float));
    float* Hc      = (float*)carve((size_t)NHEADS * NNODES * CDIM * sizeof(float));
    float* agg     = (float*)carve((size_t)NHEADS * NNODES * CDIM * sizeof(float));
    float* es      = (float*)carve((size_t)NHEADS * NNODES * sizeof(float));
    float* ed      = (float*)carve((size_t)NHEADS * NNODES * sizeof(float));
    float* maxb    = (float*)carve((size_t)NHEADS * NNODES * sizeof(float));
    float* denom   = (float*)carve((size_t)NHEADS * NNODES * sizeof(float));
    float* att     = (float*)carve((size_t)NHEADS * NEDGES * sizeof(float));

    const int B = 256;
    auto G = [](int n, int b) { return (n + b - 1) / b; };

    // dense front-end
    pack_A<<<G(MT * KT * FRAG, B), B, 0, stream>>>(node_ids, emb, Af);
    pack_B<<<G(KT * NT * FRAG, B), B, 0, stream>>>(W1, Bf);
    gemm1_wmma<<<(MTB * NTB) / (B / 32), B, 0, stream>>>(Af, Bf, b1, x1);
    gemm2<<<G(NNODES, B), B, 0, stream>>>(x1, W2, b2, hbuf);

    const int segN = NHEADS * NNODES;
    const int hcN  = NHEADS * NNODES * CDIM;
    const int aeN  = NHEADS * NEDGES;

    for (int l = 0; l < NLAYERS; ++l) {
        gat_prep<<<G(NNODES, B), B, 0, stream>>>(
            hbuf, gat_W + (size_t)l * NHEADS * CDIM * CDIM,
            gat_asrc + (size_t)l * NHEADS * CDIM, gat_adst + (size_t)l * NHEADS * CDIM,
            z, es, ed);
        init_seg<<<G(segN, B), B, 0, stream>>>(maxb, denom, segN);
        edge_max<<<G(NEDGES, B), B, 0, stream>>>(edge_src, edge_dst, es, ed, maxb);
        max_fix<<<G(segN, B), B, 0, stream>>>(maxb, segN);
        edge_exp<<<G(NEDGES, B), B, 0, stream>>>(edge_src, edge_dst, es, ed, maxb, denom, att);
        att_norm<<<G(aeN, B), B, 0, stream>>>(edge_dst, denom, att);
        copy_f32<<<G(hcN, B), B, 0, stream>>>(Hc, z, hcN);
        for (int k = 0; k < KHOPS; ++k) {
            zero_f32<<<G(hcN, B), B, 0, stream>>>(agg, hcN);
            diffuse<<<G(aeN, B), B, 0, stream>>>(edge_src, edge_dst, att, Hc, agg);
            combine<<<G(hcN, B), B, 0, stream>>>(Hc, agg, z, hcN);
        }
        head_mean<<<G(NNODES * CDIM, B), B, 0, stream>>>(Hc, hbuf, (l < NLAYERS - 1) ? 1 : 0);
    }

    zero_f32<<<G(NGRAPH * CDIM, B), B, 0, stream>>>(out, NGRAPH * CDIM);
    readout<<<G(NNODES, B), B, 0, stream>>>(hbuf, graph_ids, ws_W, ws_b, out);
}
